// DiffusionDynamics_3066606650022
// MI455X (gfx1250) — matrix operand order repaired and checked by reading, not compiled, not run on hardware
//
#include <hip/hip_runtime.h>
#include <hip/hip_bf16.h>

#define STEPS 50
#define BT 4096   // B*T
#define TT 512
#define OUTD 96

#if __has_builtin(__builtin_amdgcn_global_load_async_to_lds_b128) && \
    __has_builtin(__builtin_amdgcn_s_wait_asynccnt)
#define USE_ASYNC_LDS 1
#else
#define USE_ASYNC_LDS 0
#endif

typedef __attribute__((ext_vector_type(16))) __bf16 v16bf;
typedef __attribute__((ext_vector_type(8)))  __bf16 v8bf;
typedef __attribute__((ext_vector_type(8)))  float  v8f;

static __device__ __forceinline__ __bf16 f2bf(float f) {
  unsigned u = __builtin_bit_cast(unsigned, f);
  unsigned r = u + 0x7FFFu + ((u >> 16) & 1u);
  return __builtin_bit_cast(__bf16, (unsigned short)(r >> 16));
}
static __device__ __forceinline__ float sigm_(float x) { return 1.0f / (1.0f + __expf(-x)); }
static __device__ __forceinline__ float silu_(float x) { return x * sigm_(x); }
static __device__ __forceinline__ v16bf cat16(v8bf lo, v8bf hi) {
  return __builtin_shufflevector(lo, hi, 0,1,2,3,4,5,6,7,8,9,10,11,12,13,14,15);
}
static __device__ __forceinline__ unsigned hashu(unsigned a, unsigned b) {
  unsigned x = a * 0x9E3779B9u ^ (b + 0x7F4A7C15u);
  x ^= x >> 16; x *= 0x85EBCA6Bu; x ^= x >> 13; x *= 0xC2B2AE35u; x ^= x >> 16;
  return x;
}
static __device__ __forceinline__ float gaussr(unsigned s, unsigned i) {
  unsigned u1 = hashu(s, 2u * i + 1u), u2 = hashu(s ^ 0xB5297A4Du, 2u * i + 2u);
  float f1 = ((float)u1 + 1.0f) * (1.0f / 4294967296.0f);
  float f2 = (float)u2 * (1.0f / 4294967296.0f);
  return sqrtf(-2.0f * logf(f1)) * cosf(6.28318530718f * f2);
}

#if USE_ASYNC_LDS
// Direct global->LDS async copy (ASYNCcnt path). Toolchain signature (from
// probe diagnostic): (int4 addrspace(1)*, int4 addrspace(3)*, imm, imm).
// Generic LDS address low 32 bits equal the LDS byte offset;
// generic global address equals the global address.
typedef int v4i_ __attribute__((vector_size(4 * sizeof(int))));
typedef __attribute__((address_space(1))) v4i_* as1_v4i;
typedef __attribute__((address_space(3))) v4i_* as3_v4i;
static __device__ __forceinline__ void async_copy16(void* lds, const void* g) {
  __builtin_amdgcn_global_load_async_to_lds_b128(
      (as1_v4i)(unsigned long long)(size_t)g,
      (as3_v4i)(unsigned)(size_t)lds,
      0, 0);
}
#endif

// ---------------- generic tiled bf16 WMMA GEMM: out = act(A[M,K] @ W[N,K]^T + bias) ---
__global__ __launch_bounds__(256) void gemm_bf16(
    const __bf16* __restrict__ A, const __bf16* __restrict__ W,
    const float* __restrict__ bias, float* __restrict__ outF,
    __bf16* __restrict__ outB, int M, int N, int K, int act) {
  __shared__ __bf16 As[128][40];
  __shared__ __bf16 Bs[64][40];
  const int tid = threadIdx.x;
  const int lane = tid & 31, wid = tid >> 5;
  const int wm = wid & 3, wn = wid >> 2;       // 4x2 waves -> 128x64 block tile
  const int m0 = blockIdx.y * 128;
  const int n0 = blockIdx.x * 64;
  const int arow = lane & 15;
  const int kA = (lane >> 4) ? 8 : 0;          // A frag K interleave per ISA layout
  const int kB = (lane >> 4) * 8;              // B frag K interleave per ISA layout
  v8f acc[2][2] = {};
  for (int k0 = 0; k0 < K; k0 += 32) {
#if USE_ASYNC_LDS
    #pragma unroll
    for (int r = 0; r < 2; ++r) {              // A tile: 128x32, 16B chunks
      int ch = tid + 256 * r;
      int row = ch >> 2, c = (ch & 3) * 8;
      async_copy16(&As[row][c], A + (size_t)(m0 + row) * K + k0 + c);
    }
    {                                           // B tile: 64x32 of W rows (N guard)
      int row = tid >> 2, c = (tid & 3) * 8;
      int wr = n0 + row;
      if (wr < N) {
        async_copy16(&Bs[row][c], W + (size_t)wr * K + k0 + c);
      } else {
        v8bf bv;
        for (int i = 0; i < 8; ++i) bv[i] = f2bf(0.0f);
        *(v8bf*)&Bs[row][c] = bv;
      }
    }
    __builtin_amdgcn_s_wait_asynccnt(0);
    __syncthreads();
#else
    #pragma unroll
    for (int r = 0; r < 2; ++r) {              // A tile: 128x32, 16B chunks
      int ch = tid + 256 * r;
      int row = ch >> 2, c = (ch & 3) * 8;
      *(v8bf*)&As[row][c] = *(const v8bf*)(A + (size_t)(m0 + row) * K + k0 + c);
    }
    {                                           // B tile: 64x32 of W rows (N guard)
      int row = tid >> 2, c = (tid & 3) * 8;
      int wr = n0 + row;
      v8bf bv;
      if (wr < N) {
        bv = *(const v8bf*)(W + (size_t)wr * K + k0 + c);
      } else {
        for (int i = 0; i < 8; ++i) bv[i] = f2bf(0.0f);
      }
      *(v8bf*)&Bs[row][c] = bv;
    }
    __syncthreads();
#endif
    v16bf af[2], bfv[2];
    #pragma unroll
    for (int tm = 0; tm < 2; ++tm) {
      int r = wm * 32 + tm * 16 + arow;
      af[tm] = cat16(*(const v8bf*)&As[r][kA], *(const v8bf*)&As[r][kA + 16]);
    }
    #pragma unroll
    for (int tn = 0; tn < 2; ++tn) {
      int r = wn * 32 + tn * 16 + arow;
      bfv[tn] = cat16(*(const v8bf*)&Bs[r][kB], *(const v8bf*)&Bs[r][kB + 16]);
    }
    #pragma unroll
    for (int tm = 0; tm < 2; ++tm)
      #pragma unroll
      for (int tn = 0; tn < 2; ++tn)
        acc[tm][tn] = __builtin_amdgcn_wmma_f32_16x16x32_bf16(
            false, af[tm], false, bfv[tn], (short)0, acc[tm][tn], false, false);
    __syncthreads();
  }
  const bool hasF = outF != nullptr, hasB = outB != nullptr;
  #pragma unroll
  for (int tm = 0; tm < 2; ++tm)
    #pragma unroll
    for (int tn = 0; tn < 2; ++tn) {
      int col = n0 + wn * 32 + tn * 16 + (lane & 15);
      if (col >= N) continue;
      float bv = bias ? bias[col] : 0.0f;
      int rbase = m0 + wm * 32 + tm * 16 + ((lane >> 4) ? 8 : 0);
      #pragma unroll
      for (int i = 0; i < 8; ++i) {
        float v = acc[tm][tn][i] + bv;
        if (act == 1) v = silu_(v);
        size_t o = (size_t)(rbase + i) * N + col;
        if (hasF) outF[o] = v;
        if (hasB) outB[o] = f2bf(v);
      }
    }
}

// ---------------- sequential GRU scan; recurrent matmul h@Whh^T via WMMA -------------
__global__ __launch_bounds__(512) void gru_scan(
    const float* __restrict__ Xg,   // [B,T,768] precomputed x-side gates
    const __bf16* __restrict__ Whh, // [768,256] bf16
    const float* __restrict__ bhh,  // [768]
    float* __restrict__ outF, __bf16* __restrict__ outB) {
  __shared__ float hf[8][256];
  __shared__ __bf16 hb[16][256];   // rows 8..15 stay zero (M padding)
  __shared__ float gh[8][768];
  const int tid = threadIdx.x, lane = tid & 31, wid = tid >> 5;
  for (int i = tid; i < 16 * 256; i += 512) {
    hb[i >> 8][i & 255] = f2bf(0.0f);
    if (i < 8 * 256) hf[i >> 8][i & 255] = 0.0f;
  }
  __syncthreads();
  const int arow = lane & 15;
  const int kA = (lane >> 4) ? 8 : 0;
  const int kB = (lane >> 4) * 8;
  for (int t = 0; t < TT; ++t) {
    #pragma unroll
    for (int tt = 0; tt < 3; ++tt) {           // 48 N-tiles over 16 waves
      int n0 = (wid * 3 + tt) * 16;
      int col = n0 + arow;
      const __bf16* wrow = Whh + (size_t)col * 256 + kB;
      v8f acc = {};
      #pragma unroll
      for (int ks = 0; ks < 8; ++ks) {
        int kk = ks * 32;
        v16bf af = cat16(*(const v8bf*)&hb[arow][kk + kA],
                         *(const v8bf*)&hb[arow][kk + kA + 16]);
        v16bf bf = cat16(*(const v8bf*)(wrow + kk), *(const v8bf*)(wrow + kk + 16));
        acc = __builtin_amdgcn_wmma_f32_16x16x32_bf16(
            false, af, false, bf, (short)0, acc, false, false);
      }
      if (lane < 16) {
        #pragma unroll
        for (int i = 0; i < 8; ++i) gh[i][n0 + lane] = acc[i];
      }
    }
    __syncthreads();
    for (int idx = tid; idx < 2048; idx += 512) {
      int b = idx >> 8, j = idx & 255;
      size_t base = ((size_t)b * TT + t) * 768;
      float ir = Xg[base + j], iz = Xg[base + 256 + j], in_ = Xg[base + 512 + j];
      if (t + 1 < TT) __builtin_prefetch(Xg + base + 768 + j, 0, 1);
      float r = sigm_(ir + gh[b][j] + bhh[j]);
      float z = sigm_(iz + gh[b][256 + j] + bhh[256 + j]);
      float n = tanhf(in_ + r * (gh[b][512 + j] + bhh[512 + j]));
      float hnew = (1.0f - z) * n + z * hf[b][j];
      hf[b][j] = hnew;
      hb[b][j] = f2bf(hnew);
      size_t oo = ((size_t)b * TT + t) * 256 + j;
      if (outF) outF[oo] = hnew;
      if (outB) outB[oo] = f2bf(hnew);
    }
    __syncthreads();
  }
}

// ---------------- LayerNorm (wave per token) -----------------------------------------
__global__ __launch_bounds__(256) void layernorm_k(
    const float* __restrict__ x, const float* __restrict__ g,
    const float* __restrict__ b, __bf16* __restrict__ outB) {
  int w = threadIdx.x >> 5, lane = threadIdx.x & 31;
  int tok = blockIdx.x * 8 + w;
  const float* xr = x + (size_t)tok * 256;
  float v[8], s = 0.0f;
  #pragma unroll
  for (int i = 0; i < 8; ++i) { v[i] = xr[lane + 32 * i]; s += v[i]; }
  for (int m = 16; m; m >>= 1) s += __shfl_xor(s, m, 32);
  float mu = s * (1.0f / 256.0f), q = 0.0f;
  #pragma unroll
  for (int i = 0; i < 8; ++i) { float d = v[i] - mu; q += d * d; }
  for (int m = 16; m; m >>= 1) q += __shfl_xor(q, m, 32);
  float inv = rsqrtf(q * (1.0f / 256.0f) + 1e-5f);
  #pragma unroll
  for (int i = 0; i < 8; ++i) {
    int c = lane + 32 * i;
    outB[(size_t)tok * 256 + c] = f2bf((v[i] - mu) * inv * g[c] + b[c]);
  }
}

// ---------------- small kernels ------------------------------------------------------
__global__ void cvt_k(const float* __restrict__ s, __bf16* __restrict__ d, int n) {
  int i = blockIdx.x * blockDim.x + threadIdx.x;
  if (i < n) d[i] = f2bf(s[i]);
}
__global__ void xcat_k(const float* __restrict__ z, const float* __restrict__ o,
                       __bf16* __restrict__ d) {
  int i = blockIdx.x * blockDim.x + threadIdx.x;
  if (i < BT * 96) {
    int tok = i / 96, c = i % 96;
    d[i] = f2bf(c < 64 ? z[tok * 64 + c] : o[tok * 32 + (c - 64)]);
  }
}
__global__ void schedule_k(float* sra, float* coeff, float* sqb) {
  if (threadIdx.x == 0 && blockIdx.x == 0) {
    float ab = 1.0f;
    for (int i = 0; i < STEPS; ++i) {
      float beta = 1e-4f + (0.02f - 1e-4f) * (float)i / (float)(STEPS - 1);
      float alpha = 1.0f - beta;
      ab *= alpha;
      sra[i] = rsqrtf(alpha);
      coeff[i] = beta * rsqrtf(1.0f - ab);
      sqb[i] = sqrtf(beta);
    }
  }
}
__global__ __launch_bounds__(256) void time_emb_k(
    const float* __restrict__ w1, const float* __restrict__ b1,
    const float* __restrict__ w2, const float* __restrict__ b2,
    __bf16* __restrict__ temb) {
  __shared__ float e[256];
  int s = blockIdx.x, d = threadIdx.x;
  float tn = (float)s / (float)(STEPS - 1);
  e[d] = silu_(tn * w1[d] + b1[d]);
  __syncthreads();
  float a = b2[d];
  for (int k = 0; k < 256; ++k) a += e[k] * w2[d * 256 + k];
  temb[s * 256 + d] = f2bf(a);
}
__global__ void init_x_k(float* x) {
  int i = blockIdx.x * blockDim.x + threadIdx.x;
  if (i < BT * OUTD) x[i] = gaussr(STEPS, (unsigned)i);
}
__global__ void init_inp_k(const __bf16* __restrict__ hc, __bf16* __restrict__ inp) {
  int i = blockIdx.x * blockDim.x + threadIdx.x;
  if (i < BT * 256) {
    int tok = i >> 8, j = i & 255;
    inp[(size_t)tok * 608 + 96 + j] = hc[i];
  }
}
__global__ void update_inp_k(const float* __restrict__ x, const __bf16* __restrict__ temb,
                             int step, __bf16* __restrict__ inp) {
  int i = blockIdx.x * blockDim.x + threadIdx.x;
  int tok = i / 352, c = i % 352;
  if (tok < BT) {
    if (c < 96) inp[(size_t)tok * 608 + c] = f2bf(x[tok * 96 + c]);
    else        inp[(size_t)tok * 608 + 256 + c] = temb[step * 256 + (c - 96)];
  }
}
__global__ void diff_update_k(const float* __restrict__ xin, const float* __restrict__ eps,
                              const float* __restrict__ sra, const float* __restrict__ coeff,
                              const float* __restrict__ sqb, int step,
                              float* __restrict__ xout) {
  int i = blockIdx.x * blockDim.x + threadIdx.x;
  if (i < BT * OUTD) {
    float mean = sra[step] * (xin[i] - coeff[step] * eps[i]);
    if (step > 0) mean += sqb[step] * gaussr((unsigned)step, (unsigned)i);
    xout[i] = mean;
  }
}
__global__ void split_k(const float* __restrict__ x, float* __restrict__ out) {
  int i = blockIdx.x * blockDim.x + threadIdx.x;
  if (i < BT * OUTD) {
    int tok = i / 96, c = i % 96;
    if (c < 64) out[tok * 64 + c] = x[i];
    else        out[BT * 64 + tok * 32 + (c - 64)] = x[i];
  }
}

// ---------------- orchestration ------------------------------------------------------
extern "C" void kernel_launch(void* const* d_in, const int* in_sizes, int n_in,
                              void* d_out, int out_size, void* d_ws, size_t ws_size,
                              hipStream_t stream) {
  const float* z_seq  = (const float*)d_in[0];
  const float* o_seq  = (const float*)d_in[1];
  const float* ipw    = (const float*)d_in[2];
  const float* ipb    = (const float*)d_in[3];
  const float* wih0   = (const float*)d_in[4];
  const float* whh0   = (const float*)d_in[5];
  const float* bih0   = (const float*)d_in[6];
  const float* bhh0   = (const float*)d_in[7];
  const float* wih1   = (const float*)d_in[8];
  const float* whh1   = (const float*)d_in[9];
  const float* bih1   = (const float*)d_in[10];
  const float* bhh1   = (const float*)d_in[11];
  const float* ln_g   = (const float*)d_in[12];
  const float* ln_b   = (const float*)d_in[13];
  const float* tw1    = (const float*)d_in[14];
  const float* tb1    = (const float*)d_in[15];
  const float* tw2    = (const float*)d_in[16];
  const float* tb2    = (const float*)d_in[17];
  const float* dw1    = (const float*)d_in[18];
  const float* db1    = (const float*)d_in[19];
  const float* dw2    = (const float*)d_in[20];
  const float* db2    = (const float*)d_in[21];
  const float* dw3    = (const float*)d_in[22];
  const float* db3    = (const float*)d_in[23];

  char* ws = (char*)d_ws;
  size_t off = 0;
  auto alloc = [&](size_t bytes) -> void* {
    void* p = ws + off;
    off = (off + bytes + 255) & ~(size_t)255;
    return p;
  };
  __bf16* xcatB = (__bf16*)alloc((size_t)BT * 96 * 2);
  __bf16* ipwB  = (__bf16*)alloc((size_t)256 * 96 * 2);
  __bf16* wih0B = (__bf16*)alloc((size_t)768 * 256 * 2);
  __bf16* whh0B = (__bf16*)alloc((size_t)768 * 256 * 2);
  __bf16* wih1B = (__bf16*)alloc((size_t)768 * 256 * 2);
  __bf16* whh1B = (__bf16*)alloc((size_t)768 * 256 * 2);
  __bf16* dw1B  = (__bf16*)alloc((size_t)512 * 608 * 2);
  __bf16* dw2B  = (__bf16*)alloc((size_t)512 * 512 * 2);
  __bf16* dw3B  = (__bf16*)alloc((size_t)96 * 512 * 2);
  __bf16* hc0B  = (__bf16*)alloc((size_t)BT * 256 * 2);
  float*  XgF   = (float*)alloc((size_t)BT * 768 * 4);
  __bf16* h0B   = (__bf16*)alloc((size_t)BT * 256 * 2);
  float*  h1F   = (float*)alloc((size_t)BT * 256 * 4);
  __bf16* hcB   = (__bf16*)alloc((size_t)BT * 256 * 2);
  float*  sra   = (float*)alloc(STEPS * 4);
  float*  coeff = (float*)alloc(STEPS * 4);
  float*  sqb   = (float*)alloc(STEPS * 4);
  __bf16* tembB = (__bf16*)alloc((size_t)STEPS * 256 * 2);
  __bf16* inpB  = (__bf16*)alloc((size_t)BT * 608 * 2);
  __bf16* y1B   = (__bf16*)alloc((size_t)BT * 512 * 2);
  __bf16* y2B   = (__bf16*)alloc((size_t)BT * 512 * 2);
  float*  epsF  = (float*)alloc((size_t)BT * 96 * 4);
  float*  xA    = (float*)alloc((size_t)BT * 96 * 4);
  float*  xB    = (float*)alloc((size_t)BT * 96 * 4);

  auto cvt = [&](const float* s, __bf16* d, int n) {
    cvt_k<<<(n + 255) / 256, 256, 0, stream>>>(s, d, n);
  };
  cvt(ipw, ipwB, 256 * 96);
  cvt(wih0, wih0B, 768 * 256);
  cvt(whh0, whh0B, 768 * 256);
  cvt(wih1, wih1B, 768 * 256);
  cvt(whh1, whh1B, 768 * 256);
  cvt(dw1, dw1B, 512 * 608);
  cvt(dw2, dw2B, 512 * 512);
  cvt(dw3, dw3B, 96 * 512);

  auto gemm = [&](const __bf16* A, const __bf16* W, const float* bias,
                  float* oF, __bf16* oB, int M, int N, int K, int act) {
    dim3 g((N + 63) / 64, M / 128);
    gemm_bf16<<<g, 256, 0, stream>>>(A, W, bias, oF, oB, M, N, K, act);
  };

  // conditioning
  xcat_k<<<(BT * 96 + 255) / 256, 256, 0, stream>>>(z_seq, o_seq, xcatB);
  gemm(xcatB, ipwB, ipb, nullptr, hc0B, BT, 256, 96, 0);      // in_proj
  gemm(hc0B, wih0B, bih0, XgF, nullptr, BT, 768, 256, 0);     // x-gates layer 0
  gru_scan<<<1, 512, 0, stream>>>(XgF, whh0B, bhh0, nullptr, h0B);
  gemm(h0B, wih1B, bih1, XgF, nullptr, BT, 768, 256, 0);      // x-gates layer 1
  gru_scan<<<1, 512, 0, stream>>>(XgF, whh1B, bhh1, h1F, nullptr);
  layernorm_k<<<BT / 8, 256, 0, stream>>>(h1F, ln_g, ln_b, hcB);

  // schedule + time embeddings + x0
  schedule_k<<<1, 64, 0, stream>>>(sra, coeff, sqb);
  time_emb_k<<<STEPS, 256, 0, stream>>>(tw1, tb1, tw2, tb2, tembB);
  init_x_k<<<(BT * OUTD + 255) / 256, 256, 0, stream>>>(xA);
  init_inp_k<<<(BT * 256 + 255) / 256, 256, 0, stream>>>(hcB, inpB);

  // 50-step reverse diffusion
  for (int i = 0; i < STEPS; ++i) {
    int s = STEPS - 1 - i;
    float* xin = (i & 1) ? xB : xA;
    float* xout = (i & 1) ? xA : xB;
    update_inp_k<<<(BT * 352 + 255) / 256, 256, 0, stream>>>(xin, tembB, s, inpB);
    gemm(inpB, dw1B, db1, nullptr, y1B, BT, 512, 608, 1);
    gemm(y1B, dw2B, db2, nullptr, y2B, BT, 512, 512, 1);
    gemm(y2B, dw3B, db3, epsF, nullptr, BT, 96, 512, 0);
    diff_update_k<<<(BT * OUTD + 255) / 256, 256, 0, stream>>>(xin, epsF, sra, coeff,
                                                               sqb, s, xout);
  }
  // after 50 steps final state is back in xA
  split_k<<<(BT * OUTD + 255) / 256, 256, 0, stream>>>(xA, (float*)d_out);
}
